// S4Block_18047452578306
// MI455X (gfx1250) — matrix operand order repaired and checked by reading, not compiled, not run on hardware
//
#include <hip/hip_runtime.h>
#include <hip/hip_bf16.h>
#include <math.h>

// Problem dims (fixed by reference)
#define Bb 8
#define Ll 4096
#define Hh 256
#define Nn 64
#define HN (Hh * Nn)
#define Mm (Bb * Ll)          // 32768 rows into the projection GEMM

typedef __attribute__((ext_vector_type(16))) _Float16 v16h;
typedef __attribute__((ext_vector_type(8)))  _Float16 h8;
typedef __attribute__((ext_vector_type(8)))  float    v8f;
typedef __attribute__((ext_vector_type(4)))  unsigned int u32x4;
typedef __attribute__((ext_vector_type(4)))  int      i32x4;
typedef __attribute__((ext_vector_type(8)))  int      i32x8;

#if __has_builtin(__builtin_amdgcn_tensor_load_to_lds)
#define HAVE_TDM 1
#else
#define HAVE_TDM 0
#endif

// ---------------------------------------------------------------------------
// Kernel 1: ZOH discretization.  For each (h,n):
//   dt = exp(log_dt[h]); z = dt*A; w = exp(z);  f = (w-1)/A;  Ct_c = C_c * f
// Outputs wr,wi,c0r,c0i,c1r,c1i  each [H*N] f32.
// ---------------------------------------------------------------------------
__global__ __launch_bounds__(256) void s4_setup_kernel(
    const float* __restrict__ log_dt,
    const float* __restrict__ A_re, const float* __restrict__ A_im,
    const float* __restrict__ C_re, const float* __restrict__ C_im,
    float* __restrict__ wr,  float* __restrict__ wi,
    float* __restrict__ c0r, float* __restrict__ c0i,
    float* __restrict__ c1r, float* __restrict__ c1i)
{
    int idx = blockIdx.x * blockDim.x + threadIdx.x;   // [0, H*N)
    if (idx >= HN) return;
    int h = idx >> 6;                                   // / N
    float dt = __expf(log_dt[h]);
    float ar = A_re[idx], ai = A_im[idx];
    float zr = dt * ar, zi = dt * ai;
    float ez = __expf(zr);
    float wre = ez * __cosf(zi);
    float wim = ez * __sinf(zi);
    wr[idx] = wre;  wi[idx] = wim;
    // f = (w - 1) / A   (complex divide; |A|^2 >= 0.25 since ar = -0.5)
    float nr = wre - 1.0f, ni = wim;
    float d2 = ar * ar + ai * ai;
    float inv = 1.0f / d2;
    float fr = (nr * ar + ni * ai) * inv;
    float fi = (ni * ar - nr * ai) * inv;
    // Ct = C * f for both directions
    float cr0 = C_re[idx],      ci0 = C_im[idx];        // C[0,h,n]
    float cr1 = C_re[HN + idx], ci1 = C_im[HN + idx];   // C[1,h,n]
    c0r[idx] = cr0 * fr - ci0 * fi;
    c0i[idx] = cr0 * fi + ci0 * fr;
    c1r[idx] = cr1 * fr - ci1 * fi;
    c1i[idx] = cr1 * fi + ci1 * fr;
}

// ---------------------------------------------------------------------------
// Kernel 2: W_out [K=H][N=H] f32  ->  Wt [N][K] f16   (transpose + convert)
// ---------------------------------------------------------------------------
__global__ __launch_bounds__(256) void s4_wconv_kernel(
    const float* __restrict__ W, _Float16* __restrict__ Wt)
{
    int idx = blockIdx.x * blockDim.x + threadIdx.x;   // [0, H*H)
    int n = idx >> 8, k = idx & 255;
    Wt[(size_t)n * Hh + k] = (_Float16)W[(size_t)k * Hh + n];
}

// ---------------------------------------------------------------------------
// Kernel 3: bidirectional diagonal-SSM scan (mathematically == FFT conv path).
// One wave32 per (b,h); each lane owns states n = lane and lane+32.
//   fwd:  s_l = w s_{l-1} + u_l,          y_f[l] = 2 Re(C0 . s_l)
//   bwd:  r_l = u_{l+1} + w r_{l+1},      y_b[l] = 2 Re(C1 . r_l)
// ---------------------------------------------------------------------------
__global__ __launch_bounds__(32) void s4_scan_kernel(
    const float* __restrict__ x,
    const float* __restrict__ wr,  const float* __restrict__ wi,
    const float* __restrict__ c0r, const float* __restrict__ c0i,
    const float* __restrict__ c1r, const float* __restrict__ c1i,
    const float* __restrict__ Dsk,
    float* __restrict__ ytmp, _Float16* __restrict__ act)
{
    int bh = blockIdx.x;                 // [0, B*H)
    int b  = bh >> 8;
    int h  = bh & 255;
    int lane = threadIdx.x & 31;
    int i0 = h * Nn + lane;              // state n = lane
    int i1 = i0 + 32;                    // state n = lane + 32

    float wr0 = wr[i0], wi0 = wi[i0], wr1 = wr[i1], wi1 = wi[i1];
    float a0r = c0r[i0], a0i = c0i[i0], a1r = c0r[i1], a1i = c0i[i1];
    float b0r = c1r[i0], b0i = c1i[i0], b1r = c1r[i1], b1i = c1i[i1];
    float Dh = Dsk[h];

    const size_t rowBase = (size_t)b * Ll * Hh + h;

    // ---- forward pass ----
    float sr0 = 0.f, si0 = 0.f, sr1 = 0.f, si1 = 0.f;
    for (int l = 0; l < Ll; ++l) {
        float u = x[rowBase + (size_t)l * Hh];          // wave-uniform load
        float tr = wr0 * sr0 - wi0 * si0 + u;
        float ti = wr0 * si0 + wi0 * sr0;
        sr0 = tr; si0 = ti;
        tr = wr1 * sr1 - wi1 * si1 + u;
        ti = wr1 * si1 + wi1 * sr1;
        sr1 = tr; si1 = ti;
        float p = a0r * sr0 - a0i * si0 + a1r * sr1 - a1i * si1;
        p += __shfl_xor(p, 16);
        p += __shfl_xor(p, 8);
        p += __shfl_xor(p, 4);
        p += __shfl_xor(p, 2);
        p += __shfl_xor(p, 1);
        if (lane == 0) ytmp[rowBase + (size_t)l * Hh] = 2.0f * p;
    }

    // ---- backward pass + skip + GELU(erf) + f16 store ----
    sr0 = 0.f; si0 = 0.f; sr1 = 0.f; si1 = 0.f;        // r state
    for (int l = Ll - 1; l >= 0; --l) {
        float u = x[rowBase + (size_t)l * Hh];
        float p = b0r * sr0 - b0i * si0 + b1r * sr1 - b1i * si1;
        p += __shfl_xor(p, 16);
        p += __shfl_xor(p, 8);
        p += __shfl_xor(p, 4);
        p += __shfl_xor(p, 2);
        p += __shfl_xor(p, 1);
        if (lane == 0) {
            float v = ytmp[rowBase + (size_t)l * Hh] + 2.0f * p + Dh * u;
            float g = 0.5f * v * (1.0f + erff(v * 0.70710678118654752f));
            act[rowBase + (size_t)l * Hh] = (_Float16)g;
        }
        // r_{l-1} = u_l + w * r_l
        float tr = wr0 * sr0 - wi0 * si0 + u;
        float ti = wr0 * si0 + wi0 * sr0;
        sr0 = tr; si0 = ti;
        tr = wr1 * sr1 - wi1 * si1 + u;
        ti = wr1 * si1 + wi1 * sr1;
        sr1 = tr; si1 = ti;
    }
}

// ---------------------------------------------------------------------------
// Kernel 4: fused  out = LN( act @ W + b + x )  with v_wmma_f32_16x16x32_f16.
// Block: 256 thr (8 waves) computes a 32x256 output tile.  The block's
// 32x256 f16 A-tile is staged into LDS once by the Tensor Data Mover
// (tensor_load_to_lds, 2-D descriptor, data_size=2B), then all 8 waves read
// fragments with ds_load_b128.  B (Wt, L2-resident) streams via
// global_load_b128 + global_prefetch.
//
// Fragment maps (CDNA5 ISA, 16-bit A 16x32 / B 32x16, wave32):
//   A lane l: m = l&15;  K groups {kb..kb+7, kb+16..kb+23}, kb = (l<16?0:8)
//   B lane l: n = l&15;  K groups {kb..kb+7, kb+16..kb+23}, kb = (l<16?0:8)  (Wt row-major K)
//   C lane l: col = l&15; VGPR r holds row r + (l<16?0:8)
// ---------------------------------------------------------------------------
__global__ __launch_bounds__(256) void s4_gemm_ln_kernel(
    const _Float16* __restrict__ act, const _Float16* __restrict__ Wt,
    const float* __restrict__ x,     const float* __restrict__ bias,
    const float* __restrict__ gamma, const float* __restrict__ beta,
    float* __restrict__ out)
{
    __shared__ _Float16 aTile[32 * Hh];        // 16 KB staged A tile
    __shared__ float tile[32][Hh + 4];         // epilogue tile

    int tid  = threadIdx.x;
    int wave = tid >> 5;
    int lane = tid & 31;
    int mtile = wave & 1;        // 0..1
    int nquad = wave >> 1;       // 0..3
    int blockM = blockIdx.x * 32;

#if HAVE_TDM
    if (tid < 32) {
        // ---- D# group 0: count=1 | lds_addr | global_addr | type=2 ----
        unsigned long long ga =
            (unsigned long long)(uintptr_t)(act + (size_t)blockM * Hh);
        u32x4 g0;
        g0.x = 1u;                                     // count = 1 valid descriptor
        g0.y = (unsigned int)(uintptr_t)aTile;         // LDS byte address
        g0.z = (unsigned int)(ga & 0xFFFFFFFFu);       // global_addr[31:0]
        g0.w = (unsigned int)((ga >> 32) & 0x1FFFFFFu) // global_addr[56:32]
             | 0x80000000u;                            // type = 2 ("image")
        // ---- D# group 1: 2-D tile descriptor ----
        i32x8 g1;
        g1[0] = 0x00010000;                    // wg_mask=0, data_size=1 (2 bytes)
        g1[1] = (int)((Hh & 0xFFFF) << 16);    // tensor_dim0[15:0]=256
        g1[2] = (int)((Mm & 0xFFFF) << 16);    // dim0 hi=0 | tensor_dim1[15:0]=32768
        g1[3] = (int)(Hh << 16);               // dim1 hi=0 | tile_dim0=256
        g1[4] = 32;                            // tile_dim1=32, tile_dim2=0
        g1[5] = Hh;                            // tensor_dim0_stride = 256
        g1[6] = 0;
        g1[7] = 0;
        i32x4 z4 = {0, 0, 0, 0};
#if defined(__clang_major__) && (__clang_major__ >= 23)
        i32x8 z8 = {0, 0, 0, 0, 0, 0, 0, 0};
        __builtin_amdgcn_tensor_load_to_lds(g0, g1, z4, z4, z8, 0);
#else
        __builtin_amdgcn_tensor_load_to_lds(g0, g1, z4, z4, 0);
#endif
#if __has_builtin(__builtin_amdgcn_s_wait_tensorcnt)
        __builtin_amdgcn_s_wait_tensorcnt(0);
#endif
    }
    __syncthreads();
#else
    // Fallback: cooperative staged copy (no TDM builtin on this toolchain)
    {
        const h8* src = (const h8*)(act + (size_t)blockM * Hh);
        h8* dst = (h8*)aTile;
        for (int i = tid; i < 32 * Hh / 8; i += 256) dst[i] = src[i];
    }
    __syncthreads();
#endif

    int lhi  = (lane < 16) ? 0 : 1;
    const _Float16* arow = aTile + (size_t)(mtile * 16 + (lane & 15)) * Hh;

    v8f acc0 = {}, acc1 = {}, acc2 = {}, acc3 = {};

    for (int k = 0; k < Hh; k += 32) {
        if (k + 32 < Hh)                       // prefetch next B K-slab
            __builtin_prefetch(Wt + (size_t)(nquad * 64 + (lane & 15)) * Hh + k + 32, 0, 3);
        // ---- A fragment from LDS: two contiguous 16B ds loads ----
        int kbA = k + lhi * 8;
        h8 a0 = *(const h8*)(arow + kbA);
        h8 a1 = *(const h8*)(arow + kbA + 16);
        v16h af;
        #pragma unroll
        for (int i = 0; i < 8; ++i) { af[i] = a0[i]; af[8 + i] = a1[i]; }

        int kbB = k + lhi * 8;
        #pragma unroll
        for (int t = 0; t < 4; ++t) {
            int n = nquad * 64 + t * 16 + (lane & 15);
            const _Float16* brow = Wt + (size_t)n * Hh + kbB;
            h8 w0 = *(const h8*)(brow);
            h8 w1 = *(const h8*)(brow + 16);
            v16h bf;
            #pragma unroll
            for (int i = 0; i < 8; ++i) { bf[i] = w0[i]; bf[8 + i] = w1[i]; }
            v8f c = (t == 0) ? acc0 : (t == 1) ? acc1 : (t == 2) ? acc2 : acc3;
            c = __builtin_amdgcn_wmma_f32_16x16x32_f16(
                    false, af, false, bf, (short)0, c, false, false);
            if (t == 0) acc0 = c; else if (t == 1) acc1 = c;
            else if (t == 2) acc2 = c; else acc3 = c;
        }
    }

    // ---- epilogue: + bias + residual -> LDS ----
    #pragma unroll
    for (int t = 0; t < 4; ++t) {
        v8f c = (t == 0) ? acc0 : (t == 1) ? acc1 : (t == 2) ? acc2 : acc3;
        int n = nquad * 64 + t * 16 + (lane & 15);
        float bn = bias[n];
        #pragma unroll
        for (int r = 0; r < 8; ++r) {
            int m = mtile * 16 + lhi * 8 + r;          // row within 32-block
            float v = c[r] + bn + x[(size_t)(blockM + m) * Hh + n];
            tile[m][n] = v;
        }
    }
    __syncthreads();

    // ---- layernorm over H=256: 8 threads per row, 32 cols each ----
    int row = tid >> 3;          // 0..31
    int seg = tid & 7;           // 0..7
    float s = 0.f, s2 = 0.f;
    #pragma unroll 4
    for (int c = seg * 32; c < seg * 32 + 32; ++c) {
        float v = tile[row][c];
        s += v; s2 += v * v;
    }
    s  += __shfl_xor(s, 1);  s2 += __shfl_xor(s2, 1);
    s  += __shfl_xor(s, 2);  s2 += __shfl_xor(s2, 2);
    s  += __shfl_xor(s, 4);  s2 += __shfl_xor(s2, 4);
    float mean = s * (1.0f / Hh);
    float var  = s2 * (1.0f / Hh) - mean * mean;
    float inv  = rsqrtf(var + 1e-5f);
    #pragma unroll 4
    for (int c = seg * 32; c < seg * 32 + 32; ++c) {
        float v = (tile[row][c] - mean) * inv * gamma[c] + beta[c];
        out[(size_t)(blockM + row) * Hh + c] = v;
    }
}

// ---------------------------------------------------------------------------
extern "C" void kernel_launch(void* const* d_in, const int* in_sizes, int n_in,
                              void* d_out, int out_size, void* d_ws, size_t ws_size,
                              hipStream_t stream)
{
    const float* x      = (const float*)d_in[0];
    const float* log_dt = (const float*)d_in[1];
    const float* A_re   = (const float*)d_in[2];
    const float* A_im   = (const float*)d_in[3];
    const float* C_re   = (const float*)d_in[4];
    const float* C_im   = (const float*)d_in[5];
    const float* Dsk    = (const float*)d_in[6];
    const float* W_out  = (const float*)d_in[7];
    const float* b_out  = (const float*)d_in[8];
    const float* gamma  = (const float*)d_in[9];
    const float* beta   = (const float*)d_in[10];
    float* out = (float*)d_out;

    // workspace carve-up
    char* ws = (char*)d_ws;
    float* wr  = (float*)ws;                 ws += HN * sizeof(float);
    float* wi  = (float*)ws;                 ws += HN * sizeof(float);
    float* c0r = (float*)ws;                 ws += HN * sizeof(float);
    float* c0i = (float*)ws;                 ws += HN * sizeof(float);
    float* c1r = (float*)ws;                 ws += HN * sizeof(float);
    float* c1i = (float*)ws;                 ws += HN * sizeof(float);
    float* ytmp = (float*)ws;                ws += (size_t)Mm * Hh * sizeof(float);
    _Float16* act = (_Float16*)ws;           ws += (size_t)Mm * Hh * sizeof(_Float16);
    _Float16* Wt  = (_Float16*)ws;           ws += (size_t)Hh * Hh * sizeof(_Float16);

    s4_setup_kernel<<<(HN + 255) / 256, 256, 0, stream>>>(
        log_dt, A_re, A_im, C_re, C_im, wr, wi, c0r, c0i, c1r, c1i);

    s4_wconv_kernel<<<(Hh * Hh) / 256, 256, 0, stream>>>(W_out, Wt);

    s4_scan_kernel<<<Bb * Hh, 32, 0, stream>>>(
        x, wr, wi, c0r, c0i, c1r, c1i, Dsk, ytmp, act);

    s4_gemm_ln_kernel<<<Mm / 32, 256, 0, stream>>>(
        act, Wt, x, b_out, gamma, beta, out);
}
